// TopkRouter_16226386444875
// MI455X (gfx1250) — compile-verified, matrix operand order checked by reading, not run on hardware
//
#include <hip/hip_runtime.h>
#include <hip/hip_bf16.h>

typedef __attribute__((ext_vector_type(16))) __bf16 v16bf;
typedef __attribute__((ext_vector_type(8)))  float  v8f;
typedef __attribute__((ext_vector_type(4)))  float  f32x4;

#define DEMB 4096
#define NEXP 64
#define KSTEPS (DEMB / 32)     // 128 k-steps of 32
#define KW (KSTEPS / 4)        // 32 k-steps per wave (K-split by 4 waves)

// ---------------------------------------------------------------------------
// Pre-pass: convert W [64][4096] f32 -> bf16, swizzled into B-fragment layout.
// Fragment g = k-step index; tile t = expert group. Per lane L, 16 halfwords:
//   lane L<16 : col n = 16t+L,    halfword i<8 -> K=k0+i,    i>=8 -> K=k0+16+(i-8)
//   lane L>=16: col n = 16t+L-16, halfword i<8 -> K=k0+8+i,  i>=8 -> K=k0+24+(i-8)
// Flat storage: ((g*4+t)*32 + L)*16 + i  (each lane loads 32 contiguous bytes).
// ---------------------------------------------------------------------------
__global__ void swizzle_W_kernel(const float* __restrict__ W,
                                 unsigned short* __restrict__ wsw) {
    int tid = blockIdx.x * 256 + threadIdx.x;       // 0 .. 262143
    int i  = tid & 15;
    int L  = (tid >> 4) & 31;
    int t  = (tid >> 9) & 3;
    int g  = tid >> 11;
    int n  = t * 16 + (L & 15);
    int hi = L >> 4;
    int k  = g * 32 + (i < 8 ? hi * 8 + i : 16 + hi * 8 + (i - 8));
    __bf16 h = (__bf16)W[n * DEMB + k];
    wsw[tid] = __builtin_bit_cast(unsigned short, h);
}

// ---------------------------------------------------------------------------
// Router kernel: 1 block = 16 tokens; 4 waves K-split the 4096-deep reduction.
// ---------------------------------------------------------------------------
__global__ __launch_bounds__(128) void router_kernel(
    const float* __restrict__ X,            // [tokens][4096] f32
    const unsigned short* __restrict__ wsw, // swizzled bf16 W
    const float* __restrict__ bias,         // [64]
    float* __restrict__ out,                // [tokens][64]
    int* __restrict__ idx_out) {            // [tokens][2]
    __shared__ float lds[4][16 * NEXP];     // per-wave 16x64 partial tile

    const int lane   = threadIdx.x & 31;
    const int wv     = threadIdx.x >> 5;
    const int token0 = blockIdx.x * 16;
    const int hi     = lane >> 4;

    const float* arow = X + (size_t)(token0 + (lane & 15)) * DEMB;
    const int offA0 = hi * 8;
    const int offA1 = 16 + hi * 8;
    const int g0    = wv * KW;              // this wave's first k-step

    v8f acc0 = {}, acc1 = {}, acc2 = {}, acc3 = {};

    const v16bf* wfrag = (const v16bf*)wsw;
    // Preload B fragments for first k-step (double buffer)
    const v16bf* bp = wfrag + (size_t)g0 * 128 + lane;
    v16bf b0 = bp[0], b1 = bp[32], b2 = bp[64], b3 = bp[96];

    for (int ks = 0; ks < KW; ++ks) {
        const int g  = g0 + ks;
        const int k0 = g * 32;

        // A fragment: 16 f32/lane per ISA 16-bit A layout, cvt -> bf16
        f32x4 a0 = *(const f32x4*)(arow + k0 + offA0);
        f32x4 a1 = *(const f32x4*)(arow + k0 + offA0 + 4);
        f32x4 a2 = *(const f32x4*)(arow + k0 + offA1);
        f32x4 a3 = *(const f32x4*)(arow + k0 + offA1 + 4);
        __builtin_prefetch(arow + k0 + 64, 0, 3);   // stream next X chunk (near)

        // Prefetch next iteration's B fragments into independent registers
        int gn = g + 1; if (gn > KSTEPS - 1) gn = KSTEPS - 1;
        const v16bf* np = wfrag + (size_t)gn * 128 + lane;
        v16bf n0 = np[0], n1 = np[32], n2 = np[64], n3 = np[96];

        v16bf a;
        a[0]  = (__bf16)a0.x; a[1]  = (__bf16)a0.y; a[2]  = (__bf16)a0.z; a[3]  = (__bf16)a0.w;
        a[4]  = (__bf16)a1.x; a[5]  = (__bf16)a1.y; a[6]  = (__bf16)a1.z; a[7]  = (__bf16)a1.w;
        a[8]  = (__bf16)a2.x; a[9]  = (__bf16)a2.y; a[10] = (__bf16)a2.z; a[11] = (__bf16)a2.w;
        a[12] = (__bf16)a3.x; a[13] = (__bf16)a3.y; a[14] = (__bf16)a3.z; a[15] = (__bf16)a3.w;

        acc0 = __builtin_amdgcn_wmma_f32_16x16x32_bf16(false, a, false, b0, (short)0, acc0, false, false);
        acc1 = __builtin_amdgcn_wmma_f32_16x16x32_bf16(false, a, false, b1, (short)0, acc1, false, false);
        acc2 = __builtin_amdgcn_wmma_f32_16x16x32_bf16(false, a, false, b2, (short)0, acc2, false, false);
        acc3 = __builtin_amdgcn_wmma_f32_16x16x32_bf16(false, a, false, b3, (short)0, acc3, false, false);

        b0 = n0; b1 = n1; b2 = n2; b3 = n3;
    }

    // Dump partial logits: C layout -> row = j + hi*8, col = (lane&15) + 16t
    float* l = lds[wv];
    const int colb = lane & 15;
#pragma unroll
    for (int j = 0; j < 8; ++j) {
        const int row = j + hi * 8;
        l[row * NEXP + colb +  0] = acc0[j];
        l[row * NEXP + colb + 16] = acc1[j];
        l[row * NEXP + colb + 32] = acc2[j];
        l[row * NEXP + colb + 48] = acc3[j];
    }
    __syncthreads();

    // Cross-wave K reduction: 128 threads x 8 elements over the 16x64 tile
    float* l0 = lds[0];
    {
        const int tid = threadIdx.x;
#pragma unroll
        for (int r = 0; r < 8; ++r) {
            const int idx = tid + r * 128;
            l0[idx] = lds[0][idx] + lds[1][idx] + lds[2][idx] + lds[3][idx];
        }
    }
    __syncthreads();

    // Top-2 + sparse softmax: one thread per token row
    if (threadIdx.x < 16) {
        const int r = threadIdx.x;
        const float* lr = l0 + r * NEXP;
        float v0 = -__builtin_inff(), v1 = -__builtin_inff();
        int i0 = 0, i1 = 0;
#pragma unroll 4
        for (int e = 0; e < NEXP; ++e) {
            float v = lr[e] + bias[e];
            if (v > v0)      { v1 = v0; i1 = i0; v0 = v; i0 = e; }
            else if (v > v1) { v1 = v;  i1 = e; }
        }
        float ex  = __expf(v1 - v0);
        float inv = 1.0f / (1.0f + ex);
        float p0 = inv, p1 = ex * inv;

        float* orow = out + (size_t)(token0 + r) * NEXP;
        f32x4 z = {0.f, 0.f, 0.f, 0.f};
#pragma unroll
        for (int q = 0; q < 16; ++q) ((f32x4*)orow)[q] = z;
        orow[i0] = p0;       // same-wave stores to same addr stay ordered
        orow[i1] = p1;

        idx_out[(size_t)(token0 + r) * 2 + 0] = i0;
        idx_out[(size_t)(token0 + r) * 2 + 1] = i1;
    }
}

extern "C" void kernel_launch(void* const* d_in, const int* in_sizes, int n_in,
                              void* d_out, int out_size, void* d_ws, size_t ws_size,
                              hipStream_t stream) {
    const float* X    = (const float*)d_in[0];   // [4,4096,4096] f32
    const float* W    = (const float*)d_in[1];   // [64,4096] f32
    const float* bias = (const float*)d_in[2];   // [64] f32

    const int tokens = in_sizes[0] / DEMB;       // 16384

    float* out    = (float*)d_out;                       // [tokens][64] f32
    int*   idxout = (int*)(out + (size_t)tokens * NEXP); // [tokens][2] i32

    unsigned short* wsw = (unsigned short*)d_ws;         // 512 KB swizzled bf16 W

    // 1) swizzle + convert W (64*4096 = 262144 elements)
    swizzle_W_kernel<<<(NEXP * DEMB) / 256, 256, 0, stream>>>(W, wsw);

    // 2) router: one block per 16-token tile, 4 waves K-split each tile
    const int blocks = tokens / 16;              // 1024
    router_kernel<<<blocks, 128, 0, stream>>>(X, wsw, bias, out, idxout);
}
